// Decoder2_65472481460232
// MI455X (gfx1250) — matrix operand order repaired and checked by reading, not compiled, not run on hardware
//
#include <hip/hip_runtime.h>
#include <hip/hip_bf16.h>

// ---------------------------------------------------------------------------
// LSTM+attention decoder, 400 sequential steps. bf16 WMMA (v_wmma_f32_16x16x32_bf16)
// for gates GEMM (32x1024 @ 1024x1024^T) and logits GEMM (32x512 @ 512x8192^T).
// Weights pre-swizzled into WMMA B-fragment layout (one contiguous 32B load per
// lane per fragment). p_tf==1 => teacher forcing always => argmax feedback dead.
// ---------------------------------------------------------------------------

typedef __attribute__((ext_vector_type(16))) __bf16 v16bf;
typedef __attribute__((ext_vector_type(8)))  float  v8f;

#define B_   32
#define Td_  800
#define To_  400
#define A_   256
#define E_   512
#define V_   8192
#define G4A_ 1024          // 4*A
#define ZK_  1024          // E + A + A  (gates K)
#define CK_  512           // A + A      (logits K)

// workspace layout (bytes)
#define OFF_EMBT 0u                                   // bf16 E_*V_   (B-frags)
#define OFF_WB   (OFF_EMBT + 2u*E_*V_)                // bf16 ZK_*G4A_ (B-frags)
#define OFF_Z    (OFF_WB   + 2u*ZK_*G4A_)             // bf16 B_*ZK_  (A-frags)
#define OFF_ZC   (OFF_Z    + 2u*B_*ZK_)               // bf16 B_*CK_  (A-frags)
#define OFF_H    (OFF_ZC   + 2u*B_*CK_)               // f32 B_*A_
#define OFF_C    (OFF_H    + 4u*B_*A_)                // f32 B_*A_
#define OFF_CTX  (OFF_C    + 4u*B_*A_)                // f32 B_*A_

// ---------------------------------------------------------------------------
// Fragment layouts (CDNA5 ISA §7.12.2, wave32):
//  A 16x32 bf16: lane L holds row m=L&15; element e -> k = (L>=16?16:0)+e (packed here)
//  B 32x16 bf16: lane L holds col n=L&15; element e -> k = (L<16?0:16)+e (packed here; not ISA table order)
//  C/D 16x16 f32: vgpr r, lane L -> m = r + (L>=16?8:0), n = L&15
// Frag storage: [tile][lane][16 elems] contiguous => 32B per lane per frag.
// ---------------------------------------------------------------------------

// ----- one-time prep: swizzle [W_ih | W_hh] into bf16 B-frags (K=ZK_, N=G4A_)
__global__ void prep_Wb_kernel(const float* __restrict__ W_ih,
                               const float* __restrict__ W_hh,
                               __bf16* __restrict__ Wb) {
  int i = blockIdx.x * 256 + threadIdx.x;        // < 1048576
  int e = i & 15, lane = (i >> 4) & 31, nt = (i >> 9) & 63, kt = i >> 15;
  int k = kt * 32 + ((lane < 16) ? 0 : 16) + e;  // input dim: [x(512)|ctx(256)|h(256)]
  int n = nt * 16 + (lane & 15);                 // gate output 0..1023
  float val = (k < E_ + A_) ? W_ih[(size_t)n * (E_ + A_) + k]
                            : W_hh[(size_t)n * A_ + (k - (E_ + A_))];
  Wb[i] = (__bf16)val;
}

// ----- one-time prep: emb_W^T into bf16 B-frags (K=E_, N=V_)
__global__ void prep_embT_kernel(const float* __restrict__ emb_W,
                                 __bf16* __restrict__ embT) {
  size_t i = (size_t)blockIdx.x * 256 + threadIdx.x;  // < 4194304
  int e = (int)(i & 15), lane = (int)((i >> 4) & 31);
  int nt = (int)((i >> 9) & 511), kt = (int)(i >> 18);
  int k = kt * 32 + ((lane < 16) ? 0 : 16) + e;
  int v = nt * 16 + (lane & 15);
  embT[i] = (__bf16)emb_W[(size_t)v * E_ + k];
}

__global__ void zero_state_kernel(float* h, float* c, float* ctx) {
  int i = blockIdx.x * 256 + threadIdx.x;        // < 8192
  h[i] = 0.f; c[i] = 0.f; ctx[i] = 0.f;
}

// ----- per step: pack z = [emb_W[gt_t] | ctx | h] into bf16 A-frags (M=32,K=ZK_)
__global__ void pack_z_kernel(const float* __restrict__ emb_W,
                              const int* __restrict__ gt,
                              const float* __restrict__ h,
                              const float* __restrict__ ctx,
                              __bf16* __restrict__ zf, int t) {
  int i = blockIdx.x * 256 + threadIdx.x;        // < 32768
  int e = i & 15, lane = (i >> 4) & 31, kt = (i >> 9) & 31, mt = i >> 14;
  int m = mt * 16 + (lane & 15);                 // batch
  int k = kt * 32 + ((lane >= 16) ? 16 : 0) + e;
  float val;
  if (k < E_) {
    int word = gt[m * To_ + t];
    val = emb_W[(size_t)word * E_ + k];
  } else if (k < E_ + A_) {
    val = ctx[m * A_ + (k - E_)];
  } else {
    val = h[m * A_ + (k - E_ - A_)];
  }
  zf[i] = (__bf16)val;
}

// ----- per step: pack zc = [h | ctx] into bf16 A-frags (M=32,K=CK_)
__global__ void pack_zc_kernel(const float* __restrict__ h,
                               const float* __restrict__ ctx,
                               __bf16* __restrict__ zcf) {
  int i = blockIdx.x * 256 + threadIdx.x;        // < 16384
  int e = i & 15, lane = (i >> 4) & 31, kt = (i >> 9) & 15, mt = i >> 13;
  int m = mt * 16 + (lane & 15);
  int k = kt * 32 + ((lane >= 16) ? 16 : 0) + e;
  float val = (k < A_) ? h[m * A_ + k] : ctx[m * A_ + (k - A_)];
  zcf[i] = (__bf16)val;
}

// ----- per step: gates GEMM (WMMA bf16) + fused LSTM pointwise
// grid = 16 workgroups (16 a-columns each), 256 threads = 8 waves.
// wave w: mtile = w&1, gate = w>>1; output tile n0 = gate*256 + blockIdx.x*16
__global__ void gates_lstm_kernel(const __bf16* __restrict__ zf,
                                  const __bf16* __restrict__ Wb,
                                  const float* __restrict__ b_ih,
                                  const float* __restrict__ b_hh,
                                  float* __restrict__ h,
                                  float* __restrict__ c) {
  __shared__ float gl[32 * 64];                  // [batch][gate*16 + col]
  const int tid  = threadIdx.x;
  const int lane = tid & 31;
  const int w    = tid >> 5;
  const int mt   = w & 1;
  const int gidx = w >> 1;                       // 0..3 = i,f,g,o
  const int nt   = gidx * 16 + blockIdx.x;       // B-frag tile col (ntiles=64)
  v8f acc = {};
  for (int kt = 0; kt < 32; ++kt) {
    v16bf a = *(const v16bf*)(zf + (((mt * 32 + kt) * 32 + lane) << 4));
    v16bf b = *(const v16bf*)(Wb + ((((size_t)kt * 64 + nt) * 32 + lane) << 4));
    acc = __builtin_amdgcn_wmma_f32_16x16x32_bf16(false, a, false, b,
                                                  (short)0, acc, false, false);
  }
  const int n    = nt * 16 + (lane & 15);        // gate index 0..1023
  const float bias = b_ih[n] + b_hh[n];
  const int col  = lane & 15;
  const int mofs = (lane >> 4) * 8;
#pragma unroll
  for (int r = 0; r < 8; ++r) {
    int m = mt * 16 + mofs + r;
    gl[m * 64 + gidx * 16 + col] = acc[r] + bias;
  }
  __syncthreads();
  for (int i = tid; i < 512; i += 256) {         // 32 batches x 16 a-cols
    int b  = i >> 4;
    int cl = i & 15;
    int a  = blockIdx.x * 16 + cl;
    float ig = gl[b * 64 +  0 + cl];
    float fg = gl[b * 64 + 16 + cl];
    float gg = gl[b * 64 + 32 + cl];
    float og = gl[b * 64 + 48 + cl];
    float si = 1.f / (1.f + __expf(-ig));
    float sf = 1.f / (1.f + __expf(-fg));
    float so = 1.f / (1.f + __expf(-og));
    float cn = sf * c[b * A_ + a] + si * tanhf(gg);
    c[b * A_ + a] = cn;
    h[b * A_ + a] = so * tanhf(cn);
  }
}

// ----- per step: masked-softmax attention, one block per batch
__global__ void attn_kernel(const float* __restrict__ kmat,
                            const float* __restrict__ vmat,
                            const int* __restrict__ lens,
                            const float* __restrict__ h,
                            float* __restrict__ ctx) {
  const int b = blockIdx.x, tid = threadIdx.x;
  __shared__ float hS[A_];
  __shared__ float eS[Td_];
  __shared__ float red[256];
  hS[tid] = h[b * A_ + tid];
  __syncthreads();
  const int len = lens[b];
  float mymax = -3.4e38f;
  for (int tt = tid; tt < Td_; tt += 256) {
    const float* kr = kmat + ((size_t)b * Td_ + tt) * A_;
    float acc = 0.f;
    for (int a = 0; a < A_; a += 4) {
      float4 kv = *(const float4*)(kr + a);
      acc += kv.x * hS[a] + kv.y * hS[a + 1] + kv.z * hS[a + 2] + kv.w * hS[a + 3];
    }
    if (tt >= len) acc = -1e9f;
    eS[tt] = acc;
    mymax = fmaxf(mymax, acc);
  }
  red[tid] = mymax; __syncthreads();
  for (int s = 128; s > 0; s >>= 1) {
    if (tid < s) red[tid] = fmaxf(red[tid], red[tid + s]);
    __syncthreads();
  }
  const float mx = red[0];
  float mysum = 0.f;
  for (int tt = tid; tt < Td_; tt += 256) {
    float wv = __expf(eS[tt] - mx);
    eS[tt] = wv;
    mysum += wv;
  }
  __syncthreads();                               // everyone done reading red[0]
  red[tid] = mysum; __syncthreads();
  for (int s = 128; s > 0; s >>= 1) {
    if (tid < s) red[tid] += red[tid + s];
    __syncthreads();
  }
  const float inv = 1.f / red[0];
  // ctx[a] = sum_t attn[t] * v[b][t][a]  (coalesced across lanes in a)
  float acc = 0.f;
  const int a = tid;
  for (int tt = 0; tt < Td_; ++tt)
    acc += eS[tt] * vmat[((size_t)b * Td_ + tt) * A_ + a];
  ctx[b * A_ + a] = acc * inv;
}

// ----- per step: logits GEMM (WMMA bf16) + char_b bias, store (B,V,To) slice t
// grid = 128 workgroups x 8 waves; wave: mtile = w&1, ntile = blockIdx.x*4 + (w>>1)
__global__ void logits_kernel(const __bf16* __restrict__ zcf,
                              const __bf16* __restrict__ embT,
                              const float* __restrict__ char_b,
                              float* __restrict__ out, int t) {
  const int tid  = threadIdx.x;
  const int lane = tid & 31;
  const int w    = tid >> 5;
  const int mt   = w & 1;
  const int nt   = blockIdx.x * 4 + (w >> 1);    // 0..511
  v8f acc = {};
  for (int kt = 0; kt < 16; ++kt) {
    if (kt + 2 < 16)   // -> global_prefetch_b8
      __builtin_prefetch(embT + ((((size_t)(kt + 2) * 512 + nt) * 32 + lane) << 4), 0, 1);
    v16bf a = *(const v16bf*)(zcf  + (((mt * 16 + kt) * 32 + lane) << 4));
    v16bf b = *(const v16bf*)(embT + ((((size_t)kt * 512 + nt) * 32 + lane) << 4));
    acc = __builtin_amdgcn_wmma_f32_16x16x32_bf16(false, a, false, b,
                                                  (short)0, acc, false, false);
  }
  const int vcol = nt * 16 + (lane & 15);
  const float bias = char_b[vcol];
  const int mofs = (lane >> 4) * 8;
#pragma unroll
  for (int r = 0; r < 8; ++r) {
    int m = mt * 16 + mofs + r;
    out[(size_t)m * ((size_t)V_ * To_) + (size_t)vcol * To_ + t] = acc[r] + bias;
  }
}

// ---------------------------------------------------------------------------
extern "C" void kernel_launch(void* const* d_in, const int* in_sizes, int n_in,
                              void* d_out, int out_size, void* d_ws, size_t ws_size,
                              hipStream_t stream) {
  const float* kmat   = (const float*)d_in[0];
  const float* vmat   = (const float*)d_in[1];
  const int*   lens   = (const int*)  d_in[2];
  const int*   gt     = (const int*)  d_in[3];
  // d_in[4] = p_tf (==1): teacher forcing always on; argmax feedback is dead.
  const float* emb_W  = (const float*)d_in[5];
  const float* W_ih   = (const float*)d_in[6];
  const float* W_hh   = (const float*)d_in[7];
  const float* b_ih   = (const float*)d_in[8];
  const float* b_hh   = (const float*)d_in[9];
  const float* char_b = (const float*)d_in[10];
  // d_in[11] = u: unused since u_i < 1 always.
  float* out = (float*)d_out;

  char* ws = (char*)d_ws;
  __bf16* embT = (__bf16*)(ws + OFF_EMBT);
  __bf16* Wb   = (__bf16*)(ws + OFF_WB);
  __bf16* zf   = (__bf16*)(ws + OFF_Z);
  __bf16* zcf  = (__bf16*)(ws + OFF_ZC);
  float*  h    = (float*)(ws + OFF_H);
  float*  c    = (float*)(ws + OFF_C);
  float*  ctx  = (float*)(ws + OFF_CTX);

  // one-time (per launch) weight swizzle + state init
  prep_Wb_kernel  <<<  4096, 256, 0, stream>>>(W_ih, W_hh, Wb);
  prep_embT_kernel<<< 16384, 256, 0, stream>>>(emb_W, embT);
  zero_state_kernel<<<   32, 256, 0, stream>>>(h, c, ctx);

  for (int t = 0; t < To_; ++t) {
    pack_z_kernel    <<<128, 256, 0, stream>>>(emb_W, gt, h, ctx, zf, t);
    gates_lstm_kernel<<< 16, 256, 0, stream>>>(zf, Wb, b_ih, b_hh, h, c);
    attn_kernel      <<< 32, 256, 0, stream>>>(kmat, vmat, lens, h, ctx);
    pack_zc_kernel   <<< 64, 256, 0, stream>>>(h, ctx, zcf);
    logits_kernel    <<<128, 256, 0, stream>>>(zcf, embT, char_b, out, t);
  }
}
